// TrueSparseGlobalSpiking_58677843198505
// MI455X (gfx1250) — matrix operand order repaired and checked by reading, not compiled, not run on hardware
//
#include <hip/hip_runtime.h>
#include <hip/hip_bf16.h>

typedef __attribute__((ext_vector_type(16))) __bf16 v16bf;
typedef __attribute__((ext_vector_type(8)))  float  v8f;

struct U256 { uint4 lo; uint4 hi; };

static __device__ __forceinline__ __bf16 f2bf(float f) { return (__bf16)f; }

#define NPIX 1048576   // 64 * 128 * 128
#define NN   16384     // H*W

// ---------------------------------------------------------------------------
// Kernel 1: pack S0 (fp32, 64x16384) into bf16 WMMA A-operand layout, zero g.
// Apack linear index t = (((mt*512 + kblk)*32 + lane)*16 + e)
//   row = mt*16 + (lane&15)
//   koff = (e<8 ? e : e+8) + (lane>=16 ? 8 : 0);  k = kblk*32 + koff
// ---------------------------------------------------------------------------
__global__ void tsgs_pack_kernel(const float* __restrict__ S0,
                                 __bf16* __restrict__ ap,
                                 float* __restrict__ g) {
    int t = blockIdx.x * 256 + threadIdx.x;          // 0 .. 1048575
    int e    = t & 15;
    int lane = (t >> 4) & 31;
    int kblk = (t >> 9) & 511;
    int mt   = t >> 18;
    int row  = mt * 16 + (lane & 15);
    int koff = (e < 8 ? e : e + 8) + (lane >= 16 ? 8 : 0);
    int k    = kblk * 32 + koff;
    ap[t] = f2bf(S0[((size_t)row << 14) + k]);
    g[t]  = 0.0f;
}

// ---------------------------------------------------------------------------
// Kernel 2: g = S0 @ G via v_wmma_f32_16x16x32_bf16 (exact: values {0,1} and
// {0,1/8}).  Wave tile = 64 rows x 16 cols (acc = 4 x v8f).  4 waves/block
// cover 64 adjacent columns at the same K rows -> full cache-line use of the
// 1 GB matrix stream.  K split 8 ways (blockIdx.y); partials accumulated with
// global_atomic_add_f32.  K-loop unrolled x2 with ping-pong B buffers so the
// software pipeline needs no buffer-rotation moves.
// grid = (256, 8) x 128 threads  =>  8192 waves, 64 K-iters each.
// ---------------------------------------------------------------------------
__global__ void tsgs_gemm_kernel(const float* __restrict__ G,
                                 const __bf16* __restrict__ Apack,
                                 float* __restrict__ g) {
    const int lane  = threadIdx.x & 31;
    const int wid   = threadIdx.x >> 5;
    const int ntile = blockIdx.x * 4 + wid;   // 0..1023 -> 16 output columns
    const int kq    = blockIdx.y;             // 0..7    -> K eighth
    const int m0    = ntile * 16;

    const v8f zero8 = {0.f,0.f,0.f,0.f,0.f,0.f,0.f,0.f};
    v8f acc[4];
    #pragma unroll
    for (int mt = 0; mt < 4; ++mt) acc[mt] = zero8;

    const uint4* ap4 = (const uint4*)Apack;
    const int kb_beg = kq * 64, kb_end = kb_beg + 64;  // 64 iters (even)

    // Per-lane B column pointer: lane L reads G[k + 8*(L/16) + koff(s), m0 + (L&15)], koff(s) = s<8 ? s : s+8.
    // Advances by 32 rows (2 MB) per K-block.
    const float* brow = G + (((size_t)(kb_beg * 32 + ((lane >> 4) << 3))) << 14) + m0 + (lane & 15);
    const size_t  bstep = ((size_t)32 << 14);   // float units per K-block

    float4 bva[4], bvb[4];
    #pragma unroll
    for (int j = 0; j < 4; ++j) { const int kb = (j < 2 ? 4 * j : 4 * j + 8); bva[j] = make_float4(brow[(size_t)kb << 14], brow[(size_t)(kb + 1) << 14], brow[(size_t)(kb + 2) << 14], brow[(size_t)(kb + 3) << 14]); }
    brow += bstep;

    for (int kblk = kb_beg; kblk < kb_end; kblk += 2) {
        // ---- half 1: consume bva, prefetch bvb (row kblk+1) ----
        {
            U256 u[4];
            #pragma unroll
            for (int mt = 0; mt < 4; ++mt) {
                const size_t ai = (((size_t)(mt * 512 + kblk) * 32) + lane) * 2;
                u[mt].lo = ap4[ai];
                u[mt].hi = ap4[ai + 1];
            }
            v16bf b0;
            #pragma unroll
            for (int j = 0; j < 4; ++j) {
                const int base = j * 4;
                b0[base + 0] = f2bf(bva[j].x);
                b0[base + 1] = f2bf(bva[j].y);
                b0[base + 2] = f2bf(bva[j].z);
                b0[base + 3] = f2bf(bva[j].w);
            }
            #pragma unroll
            for (int j = 0; j < 4; ++j) { const int kb = (j < 2 ? 4 * j : 4 * j + 8); bvb[j] = make_float4(brow[(size_t)kb << 14], brow[(size_t)(kb + 1) << 14], brow[(size_t)(kb + 2) << 14], brow[(size_t)(kb + 3) << 14]); }   // always valid
            brow += bstep;
            #pragma unroll
            for (int mt = 0; mt < 4; ++mt) {
                v16bf a = __builtin_bit_cast(v16bf, u[mt]);
                acc[mt] = __builtin_amdgcn_wmma_f32_16x16x32_bf16(
                    false, a, false, b0, (short)0, acc[mt], false, false);
            }
        }
        // ---- half 2: consume bvb, prefetch bva (row kblk+2) ----
        {
            U256 u[4];
            #pragma unroll
            for (int mt = 0; mt < 4; ++mt) {
                const size_t ai = (((size_t)(mt * 512 + kblk + 1) * 32) + lane) * 2;
                u[mt].lo = ap4[ai];
                u[mt].hi = ap4[ai + 1];
            }
            v16bf b0;
            #pragma unroll
            for (int j = 0; j < 4; ++j) {
                const int base = j * 4;
                b0[base + 0] = f2bf(bvb[j].x);
                b0[base + 1] = f2bf(bvb[j].y);
                b0[base + 2] = f2bf(bvb[j].z);
                b0[base + 3] = f2bf(bvb[j].w);
            }
            if (kblk + 2 < kb_end) {
                #pragma unroll
                for (int j = 0; j < 4; ++j) { const int kb = (j < 2 ? 4 * j : 4 * j + 8); bva[j] = make_float4(brow[(size_t)kb << 14], brow[(size_t)(kb + 1) << 14], brow[(size_t)(kb + 2) << 14], brow[(size_t)(kb + 3) << 14]); }
                brow += bstep;
            }
            #pragma unroll
            for (int mt = 0; mt < 4; ++mt) {
                v16bf a = __builtin_bit_cast(v16bf, u[mt]);
                acc[mt] = __builtin_amdgcn_wmma_f32_16x16x32_bf16(
                    false, a, false, b0, (short)0, acc[mt], false, false);
            }
        }
    }

    // C layout: VGPR r -> M = r (+8 for lanes 16-31), N = lane&15
    const int ncol = m0 + (lane & 15);
    const int radd = (lane >= 16) ? 8 : 0;
    #pragma unroll
    for (int mt = 0; mt < 4; ++mt)
        #pragma unroll
        for (int r = 0; r < 8; ++r) {
            const int brow_ = mt * 16 + r + radd;
            unsafeAtomicAdd(&g[((size_t)brow_ << 14) + ncol], acc[mt][r]);
        }
}

// ---------------------------------------------------------------------------
// Kernel 3: fused circular 7x7 / 5x5 box convs (LDS halo tile) + spiking
// dynamics + outputs.  grid (8,8,64), block (16,16).
// ---------------------------------------------------------------------------
__global__ void tsgs_fused_kernel(const float* __restrict__ V0,
                                  const float* __restrict__ S0,
                                  const float* __restrict__ U,
                                  const float* __restrict__ exc_w,
                                  const float* __restrict__ inh_w,
                                  const float* __restrict__ g,
                                  const float* __restrict__ p_decay,
                                  const float* __restrict__ p_thr,
                                  const float* __restrict__ p_reset,
                                  const float* __restrict__ p_split,
                                  const float* __restrict__ p_exl,
                                  const float* __restrict__ p_exg,
                                  const float* __restrict__ p_inl,
                                  const float* __restrict__ p_ing,
                                  const float* __restrict__ p_drop,
                                  const float* __restrict__ p_lthr,
                                  float* __restrict__ outV,
                                  float* __restrict__ outS,
                                  float* __restrict__ outY) {
    __shared__ float tile[22 * 22];

    const int b  = blockIdx.z;
    const int y0 = blockIdx.y * 16, x0 = blockIdx.x * 16;
    const int tx = threadIdx.x, ty = threadIdx.y;
    const int tid = ty * 16 + tx;
    const size_t bb = (size_t)b << 14;

    for (int idx = tid; idx < 484; idx += 256) {
        const int ly = idx / 22, lx = idx - ly * 22;
        const int gy = (y0 + ly - 3) & 127;
        const int gx = (x0 + lx - 3) & 127;
        tile[idx] = S0[bb + (gy << 7) + gx];
    }
    __syncthreads();

    float s7 = 0.f, s5 = 0.f;
    #pragma unroll
    for (int dy = 0; dy < 7; ++dy) {
        float r7 = 0.f, r5 = 0.f;
        #pragma unroll
        for (int dx = 0; dx < 7; ++dx) {
            const float v = tile[(ty + dy) * 22 + tx + dx];
            r7 += v;
            if (dx >= 1 && dx <= 5) r5 += v;
        }
        s7 += r7;
        if (dy >= 1 && dy <= 5) s5 += r5;
    }

    const float decay = p_decay[0], thr = p_thr[0], rst = p_reset[0];
    const float split = p_split[0], exl = p_exl[0], exg = p_exg[0];
    const float inl = p_inl[0], ing = p_ing[0], drop = p_drop[0], lthr = p_lthr[0];

    const int ypix = y0 + ty, xpix = x0 + tx;
    const size_t i = bb + ((size_t)ypix << 7) + xpix;

    const float lateral = exl * (s7 * exc_w[0]) + inl * (s5 * inh_w[0])
                        + (exg + ing) * g[i];

    float V = decay * V0[i] + split * U[i];
    const bool active = V > lthr;
    if (active) V = V + (1.0f - split) * U[i] + lateral;
    V = fminf(fmaxf(V, -2.0f), 2.0f);

    float spike = (V > thr) ? 1.0f : 0.0f;
    const float f = (float)ypix + 0.7f * (float)xpix;
    const float frac = f - floorf(f);
    spike *= (frac > drop) ? 1.0f : 0.0f;

    if (spike > 0.0f) V = rst;

    float Y = fminf(fmaxf((V + 1.0f) * 0.5f, 0.0f), 1.0f) + 0.5f * spike;
    Y = fminf(Y, 1.0f);

    outV[i] = V;
    outS[i] = spike;
    outY[i] = Y;
}

// ---------------------------------------------------------------------------
extern "C" void kernel_launch(void* const* d_in, const int* in_sizes, int n_in,
                              void* d_out, int out_size, void* d_ws, size_t ws_size,
                              hipStream_t stream) {
    const float* V0    = (const float*)d_in[0];
    const float* S0    = (const float*)d_in[1];
    const float* U     = (const float*)d_in[2];
    const float* exc_w = (const float*)d_in[3];
    const float* inh_w = (const float*)d_in[4];
    const float* G     = (const float*)d_in[5];
    const float* decay = (const float*)d_in[6];
    const float* thr   = (const float*)d_in[7];
    const float* rst   = (const float*)d_in[8];
    const float* split = (const float*)d_in[9];
    const float* exl   = (const float*)d_in[10];
    const float* exg   = (const float*)d_in[11];
    const float* inl   = (const float*)d_in[12];
    const float* ing   = (const float*)d_in[13];
    const float* drop  = (const float*)d_in[14];
    const float* lthr  = (const float*)d_in[15];

    float* out = (float*)d_out;

    __bf16* ap;
    float*  g;
    if (ws_size >= (size_t)(6 * 1024 * 1024 + 256)) {
        ap = (__bf16*)d_ws;
        g  = (float*)((char*)d_ws + 2 * 1024 * 1024);
    } else {
        // Safe fallback: scratch inside d_out regions that are rewritten later.
        ap = (__bf16*)(out + NPIX);   // spikes region (rewritten by fused kernel)
        g  = out + 2 * (size_t)NPIX;  // Y region (g[i] read exactly by writer of Y[i])
    }

    tsgs_pack_kernel<<<4096, 256, 0, stream>>>(S0, ap, g);
    tsgs_gemm_kernel<<<dim3(256, 8), 128, 0, stream>>>(G, ap, g);
    tsgs_fused_kernel<<<dim3(8, 8, 64), dim3(16, 16), 0, stream>>>(
        V0, S0, U, exc_w, inh_w, g,
        decay, thr, rst, split, exl, exg, inl, ing, drop, lthr,
        out, out + NPIX, out + 2 * (size_t)NPIX);
}
